// CortexReasoner_51410758533245
// MI455X (gfx1250) — matrix operand order repaired and checked by reading, not compiled, not run on hardware
//
#include <hip/hip_runtime.h>
#include <hip/hip_bf16.h>

#define T_ 4096
#define D_ 1024
#define V_ 50257
#define R_ 32
#define NBR_ 8
#define KF_ 8
#define NH_ 4
#define HD_ 256

typedef __bf16 bf16_t;
typedef __attribute__((ext_vector_type(16))) __bf16 v16bf;
typedef __attribute__((ext_vector_type(8)))  float  v8f;

union Frag16 { uint4 u[2]; v16bf v; };
union Pack8  { __bf16 e[8]; uint4 u; };

static __device__ inline v8f vzero8() {
  v8f z = {0.f,0.f,0.f,0.f,0.f,0.f,0.f,0.f};
  return z;
}

// CDNA5 async copy: global -> LDS without touching VGPR data path (ASYNCcnt).
// LDS destination address = low 32 bits of the flat pointer (ISA aperture rule).
static __device__ inline void async_g2l_b128(const void* gsrc, void* ldst) {
  unsigned lo = (unsigned)(size_t)ldst;
  unsigned long long ga = (unsigned long long)(size_t)gsrc;
  asm volatile("global_load_async_to_lds_b128 %0, %1, off"
               :: "v"(lo), "v"(ga) : "memory");
}
static __device__ inline void async_wait0() {
  asm volatile("s_wait_asynccnt 0x0" ::: "memory");
}

// ---------------------------------------------------------------------------
// x = embed_W[tokens] + pos @ W_pos   (f32 + bf16 copies)
// ---------------------------------------------------------------------------
__global__ void k_embed(const int* __restrict__ tok, const float* __restrict__ embW,
                        const float* __restrict__ Wpos, float* __restrict__ xf,
                        bf16_t* __restrict__ xb) {
  int i = (blockIdx.x * blockDim.x + threadIdx.x) * 4;
  int t = i >> 10, d = i & (D_ - 1);
  int tk = tok[t];
  float p0 = (float)t;
  float p1 = (float)t / ((float)T_ + 1e-9f);
  const float* er = embW + (size_t)tk * D_;
#pragma unroll
  for (int j = 0; j < 4; j++) {
    float v = er[d + j] + p0 * Wpos[d + j] + p1 * Wpos[D_ + d + j];
    xf[i + j] = v;
    xb[i + j] = (bf16_t)v;
  }
}

// ---------------------------------------------------------------------------
// generic f32 -> bf16 conversion
// ---------------------------------------------------------------------------
__global__ void k_cvt(const float* __restrict__ x, bf16_t* __restrict__ y, int n) {
  int i = (blockIdx.x * blockDim.x + threadIdx.x) * 4;
  if (i + 3 < n) {
#pragma unroll
    for (int j = 0; j < 4; j++) y[i + j] = (bf16_t)x[i + j];
  }
}

// ---------------------------------------------------------------------------
// WMMA GEMM: C[M,N] = A[M,K] @ B[K,N], bf16 in, f32 accum.
// Block = 256 thr (8 waves). Tile 128x128, k-step 32.
// A staged via async global->LDS copies; B staged TRANSPOSED (register pack).
// sched_barrier splits batched fragment loads from the WMMA burst.
// ---------------------------------------------------------------------------
__global__ __launch_bounds__(256) void k_gemm(
    const bf16_t* __restrict__ A, const bf16_t* __restrict__ B,
    bf16_t* __restrict__ Cb, float* __restrict__ Cf, const float* __restrict__ resid,
    int Mm, int Nn, int Kk) {
  __shared__ bf16_t Alds[128][40];    // [m][k], pad 40 -> 16B aligned rows
  __shared__ bf16_t BldsT[128][40];   // [n][k], transposed tile

  int tid = threadIdx.x;
  int wv = tid >> 5, lane = tid & 31;
  int hl = (lane < 16) ? 0 : 1;
  int l16 = lane & 15;
  int n0 = blockIdx.x * 128, m0 = blockIdx.y * 128;

  v8f acc[8];
#pragma unroll
  for (int nb = 0; nb < 8; nb++) acc[nb] = vzero8();

  int ksteps = Kk >> 5;
  for (int kt = 0; kt < ksteps; kt++) {
    int k0 = kt << 5;
    // stage A tile: 128x32 bf16 via async copy (no VGPR data movement)
#pragma unroll
    for (int i = 0; i < 2; i++) {
      int c = tid + (i << 8);
      int row = c >> 2, c8 = (c & 3) << 3;
      async_g2l_b128(A + (size_t)(m0 + row) * Kk + k0 + c8, &Alds[row][c8]);
      if (kt + 1 < ksteps)
        __builtin_prefetch(A + (size_t)(m0 + row) * Kk + k0 + 32 + c8, 0, 1);
    }
    // stage B tile transposed: BldsT[n][k].  Coalesced u16 global reads
    // (lanes sweep consecutive n), packed b128 LDS stores.
#pragma unroll
    for (int i = 0; i < 2; i++) {
      int task = tid + (i << 8);          // 512 tasks = 128 n x 4 k-groups
      int n = task & 127, kg = (task >> 7) << 3;
      Pack8 pk;
#pragma unroll
      for (int j = 0; j < 8; j++)
        pk.e[j] = B[(size_t)(k0 + kg + j) * Nn + n0 + n];
      *(uint4*)(&BldsT[n][kg]) = pk.u;
    }
    async_wait0();
    __syncthreads();

    // A fragment (16x32): lane<16 -> K 0..7 & 16..23, lane>=16 -> K 8..15 & 24..31
    Frag16 af;
    {
      int row = (wv << 4) + l16;
      int bk = hl ? 8 : 0;
      af.u[0] = *(const uint4*)(&Alds[row][bk]);
      af.u[1] = *(const uint4*)(&Alds[row][bk + 16]);
    }
    // preload all 8 B fragments ...
    Frag16 bfrag[8];
    {
      int kb = hl ? 8 : 0;
#pragma unroll
      for (int nb = 0; nb < 8; nb++) {
        int ncol = (nb << 4) + l16;
        bfrag[nb].u[0] = *(const uint4*)(&BldsT[ncol][kb]);
        bfrag[nb].u[1] = *(const uint4*)(&BldsT[ncol][kb + 16]);
      }
    }
    // ... fence the scheduler, then fire 8 back-to-back independent WMMAs
    __builtin_amdgcn_sched_barrier(0);
#pragma unroll
    for (int nb = 0; nb < 8; nb++)
      acc[nb] = __builtin_amdgcn_wmma_f32_16x16x32_bf16(
          false, af.v, false, bfrag[nb].v, (short)0, acc[nb], false, false);
    __syncthreads();
  }

  // C layout: vgpr r, lane -> row = r + (lane<16?0:8), col = lane%16
#pragma unroll
  for (int nb = 0; nb < 8; nb++) {
#pragma unroll
    for (int r = 0; r < 8; r++) {
      int row = m0 + (wv << 4) + r + (hl << 3);
      int col = n0 + (nb << 4) + l16;
      float v = acc[nb][r];
      Cb[(size_t)row * Nn + col] = (bf16_t)v;
      if (Cf) {
        float rv = resid ? resid[(size_t)row * Nn + col] : 0.f;
        Cf[(size_t)row * Nn + col] = v + rv;
      }
    }
  }
}

// ---------------------------------------------------------------------------
// Flash attention. Block = 128 thr (4 waves), 64 query rows/block, head = blockIdx.y.
// Key tiles of 32. K staged [key][hd] via async copies (QK^T fragments contiguous);
// V staged TRANSPOSED [hd][key] (PV fragments contiguous).
// ---------------------------------------------------------------------------
__global__ __launch_bounds__(128) void k_attn(
    const bf16_t* __restrict__ Q, const bf16_t* __restrict__ K,
    const bf16_t* __restrict__ V, bf16_t* __restrict__ Oo) {
  __shared__ bf16_t Klds[32][264];
  __shared__ bf16_t VldsT[256][40];   // [hd][key]
  __shared__ float  Slds[64][33];
  __shared__ bf16_t Plds[64][32];
  __shared__ float  corr_l[64];
  __shared__ float  linv_l[64];

  int tid = threadIdx.x;
  int wv = tid >> 5, lane = tid & 31;
  int hl = (lane < 16) ? 0 : 1;
  int l16 = lane & 15;
  int qt = blockIdx.x * 64;
  int h = blockIdx.y;
  const float SCALE = 0.0625f;  // 1/sqrt(HD=256)

  // Q fragments for this wave's 16 rows, full HD=256 (8 k-blocks of 32)
  Frag16 qf[8];
  {
    int row = qt + (wv << 4) + l16;
    const bf16_t* qp = Q + (size_t)row * D_ + h * HD_;
#pragma unroll
    for (int kb = 0; kb < 8; kb++) {
      int bk = (kb << 5) + (hl << 4);
      qf[kb].u[0] = *(const uint4*)(qp + bk);
      qf[kb].u[1] = *(const uint4*)(qp + bk + 8);
    }
  }

  v8f o[16];
#pragma unroll
  for (int nb = 0; nb < 16; nb++) o[nb] = vzero8();
  float m_run = -1e30f, l_run = 0.f;

  for (int jt = 0; jt < T_ / 32; jt++) {
    int kbase = jt << 5;
    __syncthreads();
    // stage K tile 32x256 bf16 via async copies (1024 uint4, 128 thr x 8)
#pragma unroll
    for (int i = 0; i < 8; i++) {
      int c = tid + (i << 7);
      int row = c >> 5, c8 = (c & 31) << 3;
      async_g2l_b128(K + (size_t)(kbase + row) * D_ + h * HD_ + c8,
                     &Klds[row][c8]);
    }
    // stage V tile transposed: VldsT[hd][key].  1024 tasks = 256 hd x 4 kgroups.
#pragma unroll
    for (int i = 0; i < 8; i++) {
      int task = tid + (i << 7);
      int hd = task & 255, kg = (task >> 8) << 3;
      Pack8 pk;
#pragma unroll
      for (int j = 0; j < 8; j++)
        pk.e[j] = V[(size_t)(kbase + kg + j) * D_ + h * HD_ + hd];
      *(uint4*)(&VldsT[hd][kg]) = pk.u;
    }
    async_wait0();
    __syncthreads();

    // S = Q @ K^T : per wave 16x32 (2 fragments along keys)
    v8f sc0 = vzero8(), sc1 = vzero8();
#pragma unroll
    for (int kb = 0; kb < 8; kb++) {
      Frag16 b0, b1;
      int hd0 = (kb << 5) + (hl << 4);  // 16 contiguous hd per half-wave
      b0.u[0] = *(const uint4*)(&Klds[l16][hd0]);
      b0.u[1] = *(const uint4*)(&Klds[l16][hd0 + 8]);
      b1.u[0] = *(const uint4*)(&Klds[16 + l16][hd0]);
      b1.u[1] = *(const uint4*)(&Klds[16 + l16][hd0 + 8]);
      sc0 = __builtin_amdgcn_wmma_f32_16x16x32_bf16(false, qf[kb].v, false, b0.v,
                                                    (short)0, sc0, false, false);
      sc1 = __builtin_amdgcn_wmma_f32_16x16x32_bf16(false, qf[kb].v, false, b1.v,
                                                    (short)0, sc1, false, false);
    }
#pragma unroll
    for (int r = 0; r < 8; r++) {
      Slds[(wv << 4) + r + (hl << 3)][l16] = sc0[r];
      Slds[(wv << 4) + r + (hl << 3)][16 + l16] = sc1[r];
    }
    __syncthreads();

    // online softmax: thread t < 64 owns query row t
    if (tid < 64) {
      float mx = m_run;
#pragma unroll 8
      for (int j = 0; j < 32; j++) mx = fmaxf(mx, Slds[tid][j] * SCALE);
      float corr = __expf(m_run - mx);
      float sum = 0.f;
#pragma unroll 8
      for (int j = 0; j < 32; j++) {
        float p = __expf(Slds[tid][j] * SCALE - mx);
        sum += p;
        Plds[tid][j] = (bf16_t)p;
      }
      l_run = l_run * corr + sum;
      m_run = mx;
      corr_l[tid] = corr;
    }
    __syncthreads();

    // rescale O accumulators by per-row correction
    float cr[8];
#pragma unroll
    for (int r = 0; r < 8; r++) cr[r] = corr_l[(wv << 4) + r + (hl << 3)];
#pragma unroll
    for (int nb = 0; nb < 16; nb++)
#pragma unroll
      for (int r = 0; r < 8; r++) o[nb][r] *= cr[r];

    // P as A-fragment (16 rows x 32 keys)
    Frag16 pf;
    {
      int prow = (wv << 4) + l16;
      int pc = hl << 4;
      pf.u[0] = *(const uint4*)(&Plds[prow][pc]);
      pf.u[1] = *(const uint4*)(&Plds[prow][pc + 8]);
    }
    // O += P @ V  (k-dim = 32 keys, 16 hd-blocks); batch loads 4-at-a-time
#pragma unroll
    for (int g = 0; g < 4; g++) {
      Frag16 vfr[4];
      int kb = hl << 4;
#pragma unroll
      for (int q = 0; q < 4; q++) {
        int hd = (((g << 2) + q) << 4) + l16;
        vfr[q].u[0] = *(const uint4*)(&VldsT[hd][kb]);
        vfr[q].u[1] = *(const uint4*)(&VldsT[hd][kb + 8]);
      }
      __builtin_amdgcn_sched_barrier(0);
#pragma unroll
      for (int q = 0; q < 4; q++)
        o[(g << 2) + q] = __builtin_amdgcn_wmma_f32_16x16x32_bf16(
            false, pf.v, false, vfr[q].v, (short)0, o[(g << 2) + q], false, false);
    }
  }

  if (tid < 64) linv_l[tid] = 1.f / l_run;
  __syncthreads();
  float li[8];
#pragma unroll
  for (int r = 0; r < 8; r++) li[r] = linv_l[(wv << 4) + r + (hl << 3)];
#pragma unroll
  for (int nb = 0; nb < 16; nb++)
#pragma unroll
    for (int r = 0; r < 8; r++) {
      int row = qt + (wv << 4) + r + (hl << 3);
      int col = h * HD_ + (nb << 4) + l16;
      Oo[(size_t)row * D_ + col] = (bf16_t)(o[nb][r] * li[r]);
    }
}

// ---------------------------------------------------------------------------
// softmax(focus_map) weights
// ---------------------------------------------------------------------------
__global__ void k_focus(const float* __restrict__ f, float* __restrict__ w) {
  __shared__ float red[1024];
  __shared__ float mval, sval;
  int tid = threadIdx.x;
  float m = -1e30f;
  for (int t = tid; t < T_; t += 1024) m = fmaxf(m, f[t]);
  red[tid] = m; __syncthreads();
  for (int s = 512; s > 0; s >>= 1) { if (tid < s) red[tid] = fmaxf(red[tid], red[tid + s]); __syncthreads(); }
  if (!tid) mval = red[0];
  __syncthreads();
  float a = 0.f;
  for (int t = tid; t < T_; t += 1024) a += __expf(f[t] - mval);
  red[tid] = a; __syncthreads();
  for (int s = 512; s > 0; s >>= 1) { if (tid < s) red[tid] += red[tid + s]; __syncthreads(); }
  if (!tid) sval = red[0];
  __syncthreads();
  float inv = 1.f / sval;
  for (int t = tid; t < T_; t += 1024) w[t] = __expf(f[t] - mval) * inv;
}

// sensor[d] = sum_t w[t] * mix[t][d]
__global__ void k_sensor(const float* __restrict__ wt, const float* __restrict__ mix,
                         float* __restrict__ sensor) {
  int d = blockIdx.x * 256 + threadIdx.x;
  float a = 0.f;
  for (int t = 0; t < T_; t++) a += wt[t] * mix[(size_t)t * D_ + d];
  sensor[d] = a;
}

// scores = H_prev @ w_gate  (32 waves, one row each)
__global__ void k_scores(const float* __restrict__ Hp, const float* __restrict__ wg,
                         float* __restrict__ scores) {
  int wv = threadIdx.x >> 5, lane = threadIdx.x & 31;
  float a = 0.f;
  for (int d = lane; d < D_; d += 32) a += Hp[wv * D_ + d] * wg[d];
  for (int off = 16; off; off >>= 1) a += __shfl_down(a, off, 32);
  if (!lane) scores[wv] = a;
}

// top-8 + force io regions on
__global__ void k_topk(const float* __restrict__ scores, float* __restrict__ mask_f,
                       int* __restrict__ mask_i) {
  if (threadIdx.x == 0) {
    float s[R_]; bool sel[R_];
    for (int r = 0; r < R_; r++) { s[r] = scores[r]; sel[r] = false; }
    for (int it = 0; it < 8; it++) {
      int best = 0; float bv = -1e30f;
      for (int r = 0; r < R_; r++) if (!sel[r] && s[r] > bv) { bv = s[r]; best = r; }
      sel[best] = true;
    }
    sel[0] = true; sel[1] = true;
    for (int r = 0; r < R_; r++) { mask_i[r] = sel[r] ? 1 : 0; mask_f[r] = sel[r] ? 1.f : 0.f; }
  }
}

// A[r][d] = sum_n softmax(-dist)[n] * Hm[nbr[r][n]][d]
__global__ void k_routerA(const float* __restrict__ Hp, const int* __restrict__ maskprev,
                          const int* __restrict__ nbr, const float* __restrict__ coords,
                          float* __restrict__ A) {
  __shared__ float wsm[NBR_]; __shared__ float nd[NBR_]; __shared__ int ni[NBR_];
  int r = blockIdx.x, tid = threadIdx.x;
  if (tid < NBR_) {
    int idx = nbr[r * NBR_ + tid]; ni[tid] = idx;
    float dx = coords[r * 2] - coords[idx * 2];
    float dy = coords[r * 2 + 1] - coords[idx * 2 + 1];
    nd[tid] = -sqrtf(dx * dx + dy * dy);
  }
  __syncthreads();
  if (tid == 0) {
    float m = -1e30f;
    for (int n = 0; n < NBR_; n++) m = fmaxf(m, nd[n]);
    float s = 0.f;
    for (int n = 0; n < NBR_; n++) { wsm[n] = __expf(nd[n] - m); s += wsm[n]; }
    for (int n = 0; n < NBR_; n++) wsm[n] /= s;
  }
  __syncthreads();
  for (int d = tid; d < D_; d += 256) {
    float a = 0.f;
    for (int n = 0; n < NBR_; n++) {
      int idx = ni[n];
      float hv = maskprev[idx] ? Hp[(size_t)idx * D_ + d] : 0.f;
      a += wsm[n] * hv;
    }
    A[(size_t)r * D_ + d] = a;
  }
}

// M = A @ W_route  (tiny GEMM, memory bound)
__global__ void k_routerB(const float* __restrict__ A, const float* __restrict__ Wr,
                          float* __restrict__ M) {
  __shared__ float ash[D_];
  int r = blockIdx.x;
  for (int i = threadIdx.x; i < D_; i += 256) ash[i] = A[(size_t)r * D_ + i];
  __syncthreads();
  int e = blockIdx.y * 256 + threadIdx.x;
  float a = 0.f;
  for (int d = 0; d < D_; d++) a += ash[d] * Wr[(size_t)d * D_ + e];
  M[(size_t)r * D_ + e] = a;
}

// Xin = rmsnorm(M + sensor@io, gamma)
__global__ void k_xin(const float* __restrict__ M, const float* __restrict__ sensor,
                      const float* __restrict__ gamma, float* __restrict__ Xin) {
  __shared__ float red[256]; __shared__ float rinv;
  int r = blockIdx.x, tid = threadIdx.x;
  float ss = 0.f;
  for (int d = tid; d < D_; d += 256) {
    float v = M[(size_t)r * D_ + d] + ((r < 2) ? sensor[d] : 0.f);
    ss += v * v;
  }
  red[tid] = ss; __syncthreads();
  for (int s = 128; s; s >>= 1) { if (tid < s) red[tid] += red[tid + s]; __syncthreads(); }
  if (!tid) rinv = rsqrtf(red[0] / (float)D_ + 1e-6f);
  __syncthreads();
  for (int d = tid; d < D_; d += 256) {
    float v = M[(size_t)r * D_ + d] + ((r < 2) ? sensor[d] : 0.f);
    Xin[(size_t)r * D_ + d] = v * gamma[d] * rinv;
  }
}

// u = sigmoid(Xin@Wr[r] + sp*b_step) * (Xin@Wv[r])
__global__ void k_rwkv1(const float* __restrict__ Xin, const float* __restrict__ Wr,
                        const float* __restrict__ Wv, const float* __restrict__ bstep,
                        const int* __restrict__ stepk, float* __restrict__ ubuf) {
  __shared__ float xsh[D_];
  int r = blockIdx.x;
  for (int i = threadIdx.x; i < D_; i += 256) xsh[i] = Xin[(size_t)r * D_ + i];
  __syncthreads();
  int e = blockIdx.y * 256 + threadIdx.x;
  const float* wr = Wr + (size_t)r * D_ * D_;
  const float* wv = Wv + (size_t)r * D_ * D_;
  float yr = 0.f, yv = 0.f;
  for (int d = 0; d < D_; d++) {
    float x = xsh[d];
    yr += x * wr[(size_t)d * D_ + e];
    yv += x * wv[(size_t)d * D_ + e];
  }
  float sp = (float)stepk[0] / 7.0f;
  float g = 1.f / (1.f + __expf(-(yr + sp * bstep[(size_t)r * D_ + e])));
  ubuf[(size_t)r * D_ + e] = g * yv;
}

// H = mask * (u@Wo[r] + Xin)   -> writes directly to output H_cur
__global__ void k_rwkv2(const float* __restrict__ ubuf, const float* __restrict__ Wo,
                        const float* __restrict__ Xin, const int* __restrict__ maski,
                        float* __restrict__ Hout) {
  __shared__ float ush[D_];
  int r = blockIdx.x;
  for (int i = threadIdx.x; i < D_; i += 256) ush[i] = ubuf[(size_t)r * D_ + i];
  __syncthreads();
  int d = blockIdx.y * 256 + threadIdx.x;
  const float* wo = Wo + (size_t)r * D_ * D_;
  float a = 0.f;
  for (int e = 0; e < D_; e++) a += ush[e] * wo[(size_t)e * D_ + d];
  a += Xin[(size_t)r * D_ + d];
  Hout[(size_t)r * D_ + d] = maski[r] ? a : 0.f;
}

// ws_state + motor_state
__global__ void k_wsmotor(const float* __restrict__ H, const int* __restrict__ maski,
                          float* __restrict__ o_ws, float* __restrict__ o_motor) {
  int d = threadIdx.x;
  int cnt = 0; float s = 0.f;
  for (int r = 0; r < R_; r++) if (maski[r]) { cnt++; s += H[(size_t)r * D_ + d]; }
  o_ws[d] = s / (float)(cnt > 0 ? cnt : 1);
  o_motor[d] = H[D_ + d];   // region IO_M = 1
}

__global__ void k_rtd(const float* __restrict__ motor, const float* __restrict__ wrtd,
                      float* __restrict__ out) {
  __shared__ float red[1024];
  int tid = threadIdx.x;
  red[tid] = motor[tid] * wrtd[tid];
  __syncthreads();
  for (int s = 512; s; s >>= 1) { if (tid < s) red[tid] += red[tid + s]; __syncthreads(); }
  if (!tid) out[0] = red[0];
}

__global__ void k_facets(const float* __restrict__ motor, const float* __restrict__ Wf,
                         float* __restrict__ facets) {
  __shared__ float msh[D_];
  int k = blockIdx.x;
  for (int i = threadIdx.x; i < D_; i += 256) msh[i] = motor[i];
  __syncthreads();
  int e = blockIdx.y * 256 + threadIdx.x;
  const float* wf = Wf + (size_t)k * D_ * D_;
  float a = 0.f;
  for (int d = 0; d < D_; d++) a += msh[d] * wf[(size_t)d * D_ + e];
  facets[(size_t)k * D_ + e] = tanhf(a);
}

__global__ void k_pi(const float* __restrict__ motor, const float* __restrict__ wpi,
                     float* __restrict__ pi) {
  __shared__ float dots[KF_];
  int wv = threadIdx.x >> 5, lane = threadIdx.x & 31;
  if (wv < KF_) {
    float a = 0.f;
    for (int d = lane; d < D_; d += 32) a += motor[d] * wpi[(size_t)wv * D_ + d];
    for (int off = 16; off; off >>= 1) a += __shfl_down(a, off, 32);
    if (!lane) dots[wv] = a;
  }
  __syncthreads();
  if (threadIdx.x == 0) {
    float m = -1e30f;
    for (int k = 0; k < KF_; k++) m = fmaxf(m, dots[k]);
    float s = 0.f;
    for (int k = 0; k < KF_; k++) { dots[k] = __expf(dots[k] - m); s += dots[k]; }
    for (int k = 0; k < KF_; k++) pi[k] = dots[k] / s;
  }
}

__global__ void k_smix(const float* __restrict__ pi, const float* __restrict__ facets,
                       float* __restrict__ smix) {
  int e = blockIdx.x * 256 + threadIdx.x;
  float a = 0.f;
  for (int k = 0; k < KF_; k++) a += pi[k] * facets[(size_t)k * D_ + e];
  smix[e] = a;
}

__global__ void k_logits(const float* __restrict__ smix, const float* __restrict__ Wout,
                         float* __restrict__ logits) {
  __shared__ float ssh[D_];
  for (int i = threadIdx.x; i < D_; i += 256) ssh[i] = smix[i];
  __syncthreads();
  int v = blockIdx.x * 256 + threadIdx.x;
  if (v < V_) {
    float a = 0.f;
    for (int d = 0; d < D_; d++) a += ssh[d] * Wout[(size_t)d * V_ + v];
    logits[v] = a;
  }
}

// ---------------------------------------------------------------------------
extern "C" void kernel_launch(void* const* d_in, const int* in_sizes, int n_in,
                              void* d_out, int out_size, void* d_ws, size_t ws_size,
                              hipStream_t stream) {
  (void)in_sizes; (void)n_in; (void)out_size; (void)ws_size;
  const int*   tok      = (const int*)  d_in[0];
  const float* focus    = (const float*)d_in[1];
  const float* Hprev    = (const float*)d_in[2];
  const int*   maskprev = (const int*)  d_in[3];
  const int*   nbr      = (const int*)  d_in[4];
  const int*   stepk    = (const int*)  d_in[5];
  const float* embW     = (const float*)d_in[6];
  const float* Wpos     = (const float*)d_in[7];
  const float* Wq       = (const float*)d_in[8];
  const float* Wk       = (const float*)d_in[9];
  const float* Wva      = (const float*)d_in[10];
  const float* Woa      = (const float*)d_in[11];
  const float* wgate    = (const float*)d_in[12];
  const float* coords   = (const float*)d_in[13];
  const float* Wroute   = (const float*)d_in[14];
  const float* gamma    = (const float*)d_in[15];
  const float* Wr       = (const float*)d_in[16];
  const float* Wv       = (const float*)d_in[17];
  const float* Wo       = (const float*)d_in[18];
  const float* bstep    = (const float*)d_in[19];
  const float* Wf       = (const float*)d_in[20];
  const float* wpi      = (const float*)d_in[21];
  const float* Wout     = (const float*)d_in[22];
  const float* wrtd     = (const float*)d_in[23];

  // output layout (reference return order, flat f32)
  float* out      = (float*)d_out;
  float* o_H      = out;                         // 32*1024
  float* o_mask   = out + R_ * D_;               // 32
  float* o_logits = o_mask + R_;                 // 50257
  float* o_rtd    = o_logits + V_;               // 1
  float* o_ws     = o_rtd + 1;                   // 1024
  float* o_motor  = o_ws + D_;                   // 1024

  // workspace carve-out
  size_t off = 0;
  char* base = (char*)d_ws;
  auto alloc = [&](size_t bytes) -> void* {
    void* p = base + off;
    off += (bytes + 255) & ~(size_t)255;
    return p;
  };
  float*  xf     = (float*) alloc((size_t)T_ * D_ * 4);
  float*  mixf   = (float*) alloc((size_t)T_ * D_ * 4);
  bf16_t* xb     = (bf16_t*)alloc((size_t)T_ * D_ * 2);
  bf16_t* wqb    = (bf16_t*)alloc((size_t)D_ * D_ * 2);
  bf16_t* wkb    = (bf16_t*)alloc((size_t)D_ * D_ * 2);
  bf16_t* wvb    = (bf16_t*)alloc((size_t)D_ * D_ * 2);
  bf16_t* wob    = (bf16_t*)alloc((size_t)D_ * D_ * 2);
  bf16_t* qb     = (bf16_t*)alloc((size_t)T_ * D_ * 2);
  bf16_t* kbuf   = (bf16_t*)alloc((size_t)T_ * D_ * 2);
  bf16_t* vbuf   = (bf16_t*)alloc((size_t)T_ * D_ * 2);
  bf16_t* ab     = (bf16_t*)alloc((size_t)T_ * D_ * 2);
  float*  focusw = (float*) alloc((size_t)T_ * 4);
  float*  sensor = (float*) alloc((size_t)D_ * 4);
  float*  scores = (float*) alloc((size_t)R_ * 4);
  int*    maski  = (int*)   alloc((size_t)R_ * 4);
  float*  Arout  = (float*) alloc((size_t)R_ * D_ * 4);
  float*  Mbuf   = (float*) alloc((size_t)R_ * D_ * 4);
  float*  Xin    = (float*) alloc((size_t)R_ * D_ * 4);
  float*  ubuf   = (float*) alloc((size_t)R_ * D_ * 4);
  float*  facets = (float*) alloc((size_t)KF_ * D_ * 4);
  float*  piv    = (float*) alloc((size_t)KF_ * 4);
  float*  smix   = (float*) alloc((size_t)D_ * 4);

  // --- token mixer ---
  k_embed<<<(T_ * D_) / 1024, 256, 0, stream>>>(tok, embW, Wpos, xf, xb);
  k_cvt<<<(D_ * D_) / 1024, 256, 0, stream>>>(Wq, wqb, D_ * D_);
  k_cvt<<<(D_ * D_) / 1024, 256, 0, stream>>>(Wk, wkb, D_ * D_);
  k_cvt<<<(D_ * D_) / 1024, 256, 0, stream>>>(Wva, wvb, D_ * D_);
  k_cvt<<<(D_ * D_) / 1024, 256, 0, stream>>>(Woa, wob, D_ * D_);

  dim3 ggrid(D_ / 128, T_ / 128);
  k_gemm<<<ggrid, 256, 0, stream>>>(xb, wqb, qb,   (float*)nullptr, (const float*)nullptr, T_, D_, D_);
  k_gemm<<<ggrid, 256, 0, stream>>>(xb, wkb, kbuf, (float*)nullptr, (const float*)nullptr, T_, D_, D_);
  k_gemm<<<ggrid, 256, 0, stream>>>(xb, wvb, vbuf, (float*)nullptr, (const float*)nullptr, T_, D_, D_);

  k_attn<<<dim3(T_ / 64, NH_), 128, 0, stream>>>(qb, kbuf, vbuf, ab);

  // mix = attn_out @ Wo_attn + x  (bf16 scratch reuses qb; f32 -> mixf)
  k_gemm<<<ggrid, 256, 0, stream>>>(ab, wob, qb, mixf, xf, T_, D_, D_);

  k_focus<<<1, 1024, 0, stream>>>(focus, focusw);
  k_sensor<<<D_ / 256, 256, 0, stream>>>(focusw, mixf, sensor);

  // --- gate / router / region step ---
  k_scores<<<1, 1024, 0, stream>>>(Hprev, wgate, scores);
  k_topk<<<1, 32, 0, stream>>>(scores, o_mask, maski);
  k_routerA<<<R_, 256, 0, stream>>>(Hprev, maskprev, nbr, coords, Arout);
  k_routerB<<<dim3(R_, D_ / 256), 256, 0, stream>>>(Arout, Wroute, Mbuf);
  k_xin<<<R_, 256, 0, stream>>>(Mbuf, sensor, gamma, Xin);
  k_rwkv1<<<dim3(R_, D_ / 256), 256, 0, stream>>>(Xin, Wr, Wv, bstep, stepk, ubuf);
  k_rwkv2<<<dim3(R_, D_ / 256), 256, 0, stream>>>(ubuf, Wo, Xin, maski, o_H);
  k_wsmotor<<<1, 1024, 0, stream>>>(o_H, maski, o_ws, o_motor);
  k_rtd<<<1, 1024, 0, stream>>>(o_motor, wrtd, o_rtd);

  // --- heads ---
  k_facets<<<dim3(KF_, D_ / 256), 256, 0, stream>>>(o_motor, Wf, facets);
  k_pi<<<1, 256, 0, stream>>>(o_motor, wpi, piv);
  k_smix<<<D_ / 256, 256, 0, stream>>>(piv, facets, smix);
  k_logits<<<(V_ + 255) / 256, 256, 0, stream>>>(smix, Wout, o_logits);
}
